// cosPeTransformer_56264071577745
// MI455X (gfx1250) — matrix operand order repaired and checked by reading, not compiled, not run on hardware
//
#include <hip/hip_runtime.h>

#define DD 512  // feature dim

typedef __bf16 bf16_t;
typedef __attribute__((ext_vector_type(16))) __bf16 v16bf;
typedef __attribute__((ext_vector_type(8)))  __bf16 v8bf;
typedef __attribute__((ext_vector_type(8)))  float  v8f;

// ---------------------------------------------------------------------------
// WMMA fragment loaders in contiguous k order (k = 16*half + s for A and B, wave32), as WIDE loads.
//
// A 16(M)x32(K) bf16 from row-major M x K: lane's 16 elements are two
// contiguous 8-element chunks at element offsets 16*half and 16*half+8.
// -> two global/ds _load_b128 per fragment.
// ---------------------------------------------------------------------------
static __device__ __forceinline__ v16bf load_A16x32(const bf16_t* p, int ld, int lane) {
  const int m = lane & 15, half = lane >> 4;
  const bf16_t* base = p + (size_t)m * ld + 16 * half;
  v8bf lo = *(const v8bf*)(base);
  v8bf hi = *(const v8bf*)(base + 8);
  return __builtin_shufflevector(lo, hi, 0, 1, 2, 3, 4, 5, 6, 7, 8, 9, 10, 11, 12,
                                 13, 14, 15);
}

// B 32(K)x16(N) where B[k][n] = p[n*ld + k] (source stored row-major N x K):
// lane's 16 elements (K = 2j + 16*half pairs) are 16 CONTIGUOUS bf16 at
// p + n*ld + 16*half  -> one v16bf deref = two global_load_b128.
static __device__ __forceinline__ v16bf load_Bt32x16(const bf16_t* p, int ld, int lane) {
  const int n = lane & 15, half = lane >> 4;
  return *(const v16bf*)(p + (size_t)n * ld + 16 * half);
}

static __device__ __forceinline__ v8f wmma_bf16(v16bf a, v16bf b, v8f c) {
  return __builtin_amdgcn_wmma_f32_16x16x32_bf16(false, a, false, b, (short)0, c,
                                                 false, false);
}

// ---------------------------------------------------------------------------
// Kernel 0a: f32 -> bf16 conversion (grid-stride)
// ---------------------------------------------------------------------------
__global__ void k_cvt(const float* __restrict__ s, bf16_t* __restrict__ d, int n) {
  int i = blockIdx.x * blockDim.x + threadIdx.x;
  const int stride = gridDim.x * blockDim.x;
  for (; i < n; i += stride) d[i] = (bf16_t)s[i];
}

// ---------------------------------------------------------------------------
// Kernel 0b: f32 [R,C] -> transposed bf16 [C,R] (tiled via LDS, 32x32 tiles)
// ---------------------------------------------------------------------------
__global__ __launch_bounds__(256) void k_cvt_tr(const float* __restrict__ src,
                                                bf16_t* __restrict__ dst, int R,
                                                int C) {
  __shared__ bf16_t tile[32][33];
  const int c0 = blockIdx.x * 32, r0 = blockIdx.y * 32;
  const int tx = threadIdx.x & 31, ty = threadIdx.x >> 5;  // 32 x 8
#pragma unroll
  for (int i = ty; i < 32; i += 8)
    tile[i][tx] = (bf16_t)src[(size_t)(r0 + i) * C + c0 + tx];
  __syncthreads();
#pragma unroll
  for (int i = ty; i < 32; i += 8)
    dst[(size_t)(c0 + i) * R + r0 + tx] = tile[tx][i];
}

// ---------------------------------------------------------------------------
// Kernel 1: fused projection + bias + row-L2-normalize -> bf16
// One WG (128 thr / 4 waves) owns 16 full rows (16 x 512 output strip).
// Weights supplied TRANSPOSED (Wt[n][k]) so B fragments are contiguous loads.
// ---------------------------------------------------------------------------
__global__ __launch_bounds__(128) void k_proj_norm(const bf16_t* __restrict__ xh,
                                                   const bf16_t* __restrict__ Wt,
                                                   const float* __restrict__ bias,
                                                   bf16_t* __restrict__ outn) {
  __shared__ float sC[16 * DD];   // 32 KB f32 tile
  __shared__ float sred[16 * 8];
  const int tid = threadIdx.x;
  const int wave = tid >> 5, lane = tid & 31;
  const int half = lane >> 4, nlane = lane & 15;
  const int m0 = blockIdx.x * 16;
  const int n0w = wave * 128;

  v8f acc[8];
#pragma unroll
  for (int t = 0; t < 8; ++t)
#pragma unroll
    for (int e = 0; e < 8; ++e) acc[t][e] = 0.0f;

#pragma unroll 4
  for (int kk = 0; kk < 16; ++kk) {
    const int k0 = kk * 32;
    v16bf a = load_A16x32(xh + (size_t)m0 * DD + k0, DD, lane);
#pragma unroll
    for (int t = 0; t < 8; ++t) {
      v16bf b = load_Bt32x16(Wt + (size_t)(n0w + t * 16) * DD + k0, DD, lane);
      acc[t] = wmma_bf16(a, b, acc[t]);
    }
  }

  // C layout: element (r,lane) -> row r + 8*half, col nlane (+16*t)
#pragma unroll
  for (int t = 0; t < 8; ++t) {
    const int n = n0w + t * 16 + nlane;
    const float bv = bias[n];
#pragma unroll
    for (int r = 0; r < 8; ++r) sC[(r + 8 * half) * DD + n] = acc[t][r] + bv;
  }
  __syncthreads();

  // row sum of squares: 8 threads per row, 64 elements each
  {
    const int row = tid >> 3, seg = tid & 7;
    const float* rp = sC + row * DD + seg * 64;
    float s = 0.f;
#pragma unroll 8
    for (int c2 = 0; c2 < 64; ++c2) { const float v = rp[c2]; s += v * v; }
    sred[row * 8 + seg] = s;
  }
  __syncthreads();
  if (tid < 16) {
    float s = 0.f;
#pragma unroll
    for (int j = 0; j < 8; ++j) s += sred[tid * 8 + j];
    sred[tid * 8] = 1.0f / fmaxf(sqrtf(s), 1e-6f);
  }
  __syncthreads();

  for (int idx = tid; idx < 16 * DD; idx += 128) {
    const int row = idx >> 9;  // /512
    outn[(size_t)(m0 + row) * DD + (idx & (DD - 1))] = (bf16_t)(sC[idx] * sred[row * 8]);
  }
}

// ---------------------------------------------------------------------------
// Kernel 2: flash pass 1 — per-row max m_i and l_i = sum exp(a - m_i) of
// A = Kn @ Qn^T. WG = 128 thr (4 waves), each wave owns 16 rows; A-fragments
// for the whole K=512 are hoisted into registers (16 x v16bf).
// ---------------------------------------------------------------------------
__global__ __launch_bounds__(128) void k_stats(const bf16_t* __restrict__ Kn,
                                               const bf16_t* __restrict__ Qn,
                                               float* __restrict__ mOut,
                                               float* __restrict__ lOut, int Ntot) {
  const int tid = threadIdx.x;
  const int wave = tid >> 5, lane = tid & 31;
  const int half = lane >> 4;
  const int i0 = blockIdx.x * 64 + wave * 16;

  v16bf aF[16];
#pragma unroll
  for (int kk = 0; kk < 16; ++kk)
    aF[kk] = load_A16x32(Kn + (size_t)i0 * DD + kk * 32, DD, lane);

  float m[8], l[8];
#pragma unroll
  for (int r = 0; r < 8; ++r) { m[r] = -1e30f; l[r] = 0.f; }

  for (int j0 = 0; j0 < Ntot; j0 += 64) {
    v8f c[4];
#pragma unroll
    for (int t = 0; t < 4; ++t)
#pragma unroll
      for (int e = 0; e < 8; ++e) c[t][e] = 0.f;

#pragma unroll 4
    for (int kk = 0; kk < 16; ++kk) {
      const int k0 = kk * 32;
#pragma unroll
      for (int t = 0; t < 4; ++t) {
        v16bf b = load_Bt32x16(Qn + (size_t)(j0 + t * 16) * DD + k0, DD, lane);
        c[t] = wmma_bf16(aF[kk], b, c[t]);
      }
    }

    // online softmax stats; each row's 64 cols live across 16 lanes x 4 tiles
#pragma unroll
    for (int r = 0; r < 8; ++r) {
      float tmax = fmaxf(fmaxf(c[0][r], c[1][r]), fmaxf(c[2][r], c[3][r]));
#pragma unroll
      for (int off = 1; off < 16; off <<= 1)
        tmax = fmaxf(tmax, __shfl_xor(tmax, off, 32));
      const float mnew = fmaxf(m[r], tmax);
      float ssum = __expf(c[0][r] - mnew) + __expf(c[1][r] - mnew) +
                   __expf(c[2][r] - mnew) + __expf(c[3][r] - mnew);
#pragma unroll
      for (int off = 1; off < 16; off <<= 1) ssum += __shfl_xor(ssum, off, 32);
      l[r] = l[r] * __expf(m[r] - mnew) + ssum;
      m[r] = mnew;
    }
  }

  if ((lane & 15) == 0) {
#pragma unroll
    for (int r = 0; r < 8; ++r) {
      const int row = i0 + r + 8 * half;
      mOut[row] = m[r];
      lOut[row] = l[r];
    }
  }
}

// ---------------------------------------------------------------------------
// Kernel 3: flash pass 2 — P = exp(Kn@Qn^T - m), Z = (P @ x) / l.
// WG = 512 thr (16 waves), 64 rows per WG. Score tile 64x128 -> LDS (bf16,
// doubles as the C->A layout transpose), then P·V with each wave owning
// 16 rows x 128 output columns. V fragments come from x^T (contiguous loads).
// ---------------------------------------------------------------------------
__global__ __launch_bounds__(512) void k_out(const bf16_t* __restrict__ Kn,
                                             const bf16_t* __restrict__ Qn,
                                             const bf16_t* __restrict__ xt,
                                             const float* __restrict__ mIn,
                                             const float* __restrict__ lIn,
                                             float* __restrict__ out, int Ntot) {
  __shared__ __align__(32) bf16_t sP[64 * 128];  // 16 KB
  const int tid = threadIdx.x;
  const int wave = tid >> 5, lane = tid & 31;
  const int half = lane >> 4, nlane = lane & 15;
  const int rg = wave >> 2;   // row group 0..3 (16 rows each)
  const int sub = wave & 3;   // col quarter (scores: 32 cols; output: 128 cols)
  const int i0 = blockIdx.x * 64 + rg * 16;

  float mrow[8], linv[8];
#pragma unroll
  for (int r = 0; r < 8; ++r) {
    const int row = i0 + r + 8 * half;
    mrow[r] = mIn[row];
    linv[r] = 1.0f / lIn[row];
  }

  v8f acc[8];
#pragma unroll
  for (int t = 0; t < 8; ++t)
#pragma unroll
    for (int e = 0; e < 8; ++e) acc[t][e] = 0.0f;

  for (int j0 = 0; j0 < Ntot; j0 += 128) {
    // ---- scores: this wave computes 16 x 32 of the 64 x 128 tile ----
    v8f c[2];
#pragma unroll
    for (int t = 0; t < 2; ++t)
#pragma unroll
      for (int e = 0; e < 8; ++e) c[t][e] = 0.f;

#pragma unroll 4
    for (int kk = 0; kk < 16; ++kk) {
      const int k0 = kk * 32;
      v16bf a = load_A16x32(Kn + (size_t)i0 * DD + k0, DD, lane);
#pragma unroll
      for (int t = 0; t < 2; ++t) {
        v16bf b =
            load_Bt32x16(Qn + (size_t)(j0 + sub * 32 + t * 16) * DD + k0, DD, lane);
        c[t] = wmma_bf16(a, b, c[t]);
      }
    }

    __syncthreads();  // previous iteration's P-V reads must be done
#pragma unroll
    for (int t = 0; t < 2; ++t) {
      const int col = sub * 32 + t * 16 + nlane;
#pragma unroll
      for (int r = 0; r < 8; ++r)
        sP[(rg * 16 + r + 8 * half) * 128 + col] = (bf16_t)__expf(c[t][r] - mrow[r]);
    }
    __syncthreads();  // P visible to all waves

    // ---- P(16x128) @ x(128x128-slice): 4 k-steps x 8 d-tiles ----
    // B[k][n] = x[j0+k][d0+n] = xt[(d0+n)*Ntot + j0+k]  (contiguous in k)
#pragma unroll
    for (int kk = 0; kk < 4; ++kk) {
      v16bf a = load_A16x32(sP + (rg * 16) * 128 + kk * 32, 128, lane);
#pragma unroll
      for (int dt = 0; dt < 8; ++dt) {
        v16bf b = load_Bt32x16(
            xt + (size_t)(sub * 128 + dt * 16) * Ntot + j0 + kk * 32, Ntot, lane);
        acc[dt] = wmma_bf16(a, b, acc[dt]);
      }
    }
  }

#pragma unroll
  for (int dt = 0; dt < 8; ++dt) {
#pragma unroll
    for (int r = 0; r < 8; ++r) {
      const int row = i0 + r + 8 * half;
      out[(size_t)row * DD + sub * 128 + dt * 16 + nlane] = acc[dt][r] * linv[r];
    }
  }
}

// ---------------------------------------------------------------------------
extern "C" void kernel_launch(void* const* d_in, const int* in_sizes, int n_in,
                              void* d_out, int out_size, void* d_ws, size_t ws_size,
                              hipStream_t stream) {
  (void)in_sizes; (void)n_in; (void)out_size; (void)ws_size;
  const int N = 8192;
  const float* x   = (const float*)d_in[0];
  const float* Wq  = (const float*)d_in[1];
  const float* Wq0 = (const float*)d_in[2];
  const float* Wk  = (const float*)d_in[3];
  const float* Wk0 = (const float*)d_in[4];
  float* out = (float*)d_out;

  // workspace layout (~33 MB total)
  char* ws = (char*)d_ws;
  bf16_t* xh  = (bf16_t*)ws;                         // N*DD bf16   (8 MB)
  bf16_t* xt  = xh + (size_t)N * DD;                 // DD*N bf16   (8 MB, x^T)
  bf16_t* wqt = xt + (size_t)N * DD;                 // DD*DD bf16  (0.5 MB, Wq^T)
  bf16_t* wkt = wqt + (size_t)DD * DD;               // DD*DD bf16  (0.5 MB, Wk^T)
  bf16_t* Qn  = wkt + (size_t)DD * DD;               // N*DD bf16   (8 MB)
  bf16_t* Kn  = Qn + (size_t)N * DD;                 // N*DD bf16   (8 MB)
  float*  mB  = (float*)(Kn + (size_t)N * DD);       // N f32
  float*  lB  = mB + N;                              // N f32

  k_cvt<<<2048, 256, 0, stream>>>(x, xh, N * DD);
  k_cvt_tr<<<dim3(DD / 32, N / 32), 256, 0, stream>>>(x, xt, N, DD);
  k_cvt_tr<<<dim3(DD / 32, DD / 32), 256, 0, stream>>>(Wq, wqt, DD, DD);
  k_cvt_tr<<<dim3(DD / 32, DD / 32), 256, 0, stream>>>(Wk, wkt, DD, DD);

  k_proj_norm<<<N / 16, 128, 0, stream>>>(xh, wqt, Wq0, Qn);
  k_proj_norm<<<N / 16, 128, 0, stream>>>(xh, wkt, Wk0, Kn);

  k_stats<<<N / 64, 128, 0, stream>>>(Kn, Qn, mB, lB, N);
  k_out<<<N / 64, 512, 0, stream>>>(Kn, Qn, xt, mB, lB, out, N);
}
